// RNNCell_85212151153259
// MI455X (gfx1250) — matrix operand order repaired and checked by reading, not compiled, not run on hardware
//
#include <hip/hip_runtime.h>

#define RNN_N 64
#define RNN_L 512
#define RNN_H 1024
#define GBLK 16          // persistent workgroups in the scan kernel
#define SEQ_BLOCK 128    // 4 waves per scan workgroup

typedef __attribute__((ext_vector_type(16))) __bf16 v16bf;
typedef __attribute__((ext_vector_type(8)))  float  v8f;
typedef unsigned int u32x4 __attribute__((ext_vector_type(4)));
typedef int          i32x4 __attribute__((ext_vector_type(4)));
typedef int          i32x8 __attribute__((ext_vector_type(8)));

union Frag {
  v16bf v;
  unsigned short us[16];
  uint4 q[2];
};

__device__ __forceinline__ unsigned short f2bf(float f) {
  unsigned u = __float_as_uint(f);
  u += 0x7FFFu + ((u >> 16) & 1u);     // round-to-nearest-even
  return (unsigned short)(u >> 16);
}

// A fragment: 16x32 bf16, per ISA layout. half = lane>>4.
// VGPR g (us[2g],us[2g+1]) holds K = k0 + 2g + (g>=4 ? 8 : 0) + 8*half
// => two runs of 8 consecutive K: [k0+8h .. +7] and [k0+16+8h .. +7]
__device__ __forceinline__ void fill_a_frag_f32(Frag& a, const float* pa) {
  float4 f0 = *(const float4*)(pa + 0);
  float4 f1 = *(const float4*)(pa + 4);
  float4 f2 = *(const float4*)(pa + 16);
  float4 f3 = *(const float4*)(pa + 20);
  a.us[ 0]=f2bf(f0.x); a.us[ 1]=f2bf(f0.y); a.us[ 2]=f2bf(f0.z); a.us[ 3]=f2bf(f0.w);
  a.us[ 4]=f2bf(f1.x); a.us[ 5]=f2bf(f1.y); a.us[ 6]=f2bf(f1.z); a.us[ 7]=f2bf(f1.w);
  a.us[ 8]=f2bf(f2.x); a.us[ 9]=f2bf(f2.y); a.us[10]=f2bf(f2.z); a.us[11]=f2bf(f2.w);
  a.us[12]=f2bf(f3.x); a.us[13]=f2bf(f3.y); a.us[14]=f2bf(f3.z); a.us[15]=f2bf(f3.w);
}

// Same fragment, but the source is already bf16 (no conversion on the critical path)
__device__ __forceinline__ void fill_a_frag_bf16(Frag& a, const unsigned short* pa) {
  *(uint4*)&a.us[0] = *(const uint4*)(pa);        // K run 0: 8 bf16
  *(uint4*)&a.us[8] = *(const uint4*)(pa + 16);   // K run 1: 8 bf16
}

// B fragment: 32x16 bf16 (KxN), N = lane&15, per-lane K runs = [k0+8*half .. +7] and [k0+16+8*half .. +7]
__device__ __forceinline__ void fill_b_frag_bf16(Frag& b, const unsigned short* pb) {
  b.q[0] = *(const uint4*)(pb);
  b.q[1] = *(const uint4*)(pb + 16);
}

// ---------------------------------------------------------------------------
// Tensor Data Mover: 2D bf16 tile (rows x rowElems) global -> LDS (D# per ISA §8)
// Unconditional this round: proves the builtin+arity on this toolchain.
// ---------------------------------------------------------------------------
__device__ __forceinline__ void tdm_load_2d_bf16_to_lds(const void* gaddr, unsigned ldsOff,
                                                        unsigned rows, unsigned rowElems,
                                                        unsigned strideElems) {
  unsigned long long ga = (unsigned long long)gaddr;
  u32x4 g0;
  g0[0] = 1u;                                               // count=1, no gather
  g0[1] = ldsOff;                                           // lds_addr (bytes)
  g0[2] = (unsigned)(ga & 0xFFFFFFFFu);                     // global_addr[31:0]
  g0[3] = (unsigned)((ga >> 32) & 0x01FFFFFFu) | (2u << 30);// global_addr[56:32] | type=2
  i32x8 g1;
  g1[0] = (int)(1u << 16);                                  // data_size=1 (2 bytes)
  g1[1] = (int)((rowElems & 0xFFFFu) << 16);                // tensor_dim0[15:0]
  g1[2] = (int)(((rowElems >> 16) & 0xFFFFu) |
                ((rows & 0xFFFFu) << 16));                  // tensor_dim0 hi | tensor_dim1 lo
  g1[3] = (int)(((rows >> 16) & 0xFFFFu) |
                ((rowElems & 0xFFFFu) << 16));              // tensor_dim1 hi | tile_dim0
  g1[4] = (int)(rows & 0xFFFFu);                            // tile_dim1 | tile_dim2=0
  g1[5] = (int)strideElems;                                 // tensor_dim0_stride[31:0]
  g1[6] = 0;
  g1[7] = 0;
  i32x4 z4 = {0, 0, 0, 0};
#if __clang_major__ >= 23
  i32x8 z8 = {0, 0, 0, 0, 0, 0, 0, 0};
  __builtin_amdgcn_tensor_load_to_lds(g0, g1, z4, z4, z8, 0);
#else
  __builtin_amdgcn_tensor_load_to_lds(g0, g1, z4, z4, 0);
#endif
}

// ---------------------------------------------------------------------------
// Kernel 0: fp32 -> bf16 weight conversion, bias fusion, barrier reset
// ---------------------------------------------------------------------------
__global__ void prep_kernel(const float* __restrict__ Wxh, const float* __restrict__ Whh,
                            const float* __restrict__ bxh, const float* __restrict__ bhh,
                            unsigned short* __restrict__ WxhB, unsigned short* __restrict__ WhhB,
                            float* __restrict__ bias, unsigned* __restrict__ ctr) {
  int i = blockIdx.x * blockDim.x + threadIdx.x;
  const int total = RNN_H * RNN_H;
  for (int idx = i; idx < total; idx += gridDim.x * blockDim.x) {
    WxhB[idx] = f2bf(Wxh[idx]);
    WhhB[idx] = f2bf(Whh[idx]);
  }
  if (i < RNN_H) bias[i] = bxh[i] + bhh[i];
  if (i == 0) *ctr = 0u;
}

// ---------------------------------------------------------------------------
// Kernel 1: xh = emb[x] @ Wxh^T + (bxh + bhh), written into d_out (in place of h)
// ---------------------------------------------------------------------------
__global__ __launch_bounds__(256) void xh_gemm(const int* __restrict__ x,
                                               const float* __restrict__ emb,
                                               const unsigned short* __restrict__ Wb,
                                               const float* __restrict__ bias,
                                               float* __restrict__ out) {
  const int lane = threadIdx.x & 31;
  const int wave = threadIdx.x >> 5;
  const int half = lane >> 4;
  const int l16  = lane & 15;
  const int mbase = blockIdx.y * 128 + wave * 16;
  const int nbase = blockIdx.x * 64;

  const int m = mbase + l16;
  const int tok = x[m];
  const float* arow = emb + (size_t)tok * RNN_H;

  v8f acc0 = {}, acc1 = {}, acc2 = {}, acc3 = {};

  for (int k0 = 0; k0 < RNN_H; k0 += 32) {
    Frag a;
    const float* pa = arow + k0 + 8 * half;
    fill_a_frag_f32(a, pa);
    __builtin_prefetch(pa + 32, 0, 0);

    Frag b0, b1, b2, b3;
    const unsigned short* pb = Wb + (size_t)(nbase + l16) * RNN_H + k0 + 8 * half;
    fill_b_frag_bf16(b0, pb);
    fill_b_frag_bf16(b1, pb + 16 * RNN_H);
    fill_b_frag_bf16(b2, pb + 32 * RNN_H);
    fill_b_frag_bf16(b3, pb + 48 * RNN_H);

    acc0 = __builtin_amdgcn_wmma_f32_16x16x32_bf16(false, a.v, false, b0.v, (short)0, acc0, false, false);
    acc1 = __builtin_amdgcn_wmma_f32_16x16x32_bf16(false, a.v, false, b1.v, (short)0, acc1, false, false);
    acc2 = __builtin_amdgcn_wmma_f32_16x16x32_bf16(false, a.v, false, b2.v, (short)0, acc2, false, false);
    acc3 = __builtin_amdgcn_wmma_f32_16x16x32_bf16(false, a.v, false, b3.v, (short)0, acc3, false, false);
  }

  const float bb0 = bias[nbase + l16];
  const float bb1 = bias[nbase + 16 + l16];
  const float bb2 = bias[nbase + 32 + l16];
  const float bb3 = bias[nbase + 48 + l16];
#pragma unroll
  for (int r = 0; r < 8; ++r) {
    const int mm = mbase + r + 8 * half;         // C/D layout: VGPR r -> M = r + 8*half
    float* po = out + (size_t)mm * RNN_H + nbase + l16;
    po[0]  = acc0[r] + bb0;
    po[16] = acc1[r] + bb1;
    po[32] = acc2[r] + bb2;
    po[48] = acc3[r] + bb3;
  }
}

// ---------------------------------------------------------------------------
// Device-wide step barrier (monotonic counter; reset by prep_kernel each launch)
// ---------------------------------------------------------------------------
__device__ __forceinline__ void grid_sync(unsigned* ctr, unsigned target) {
  __threadfence();
  __syncthreads();
  if (threadIdx.x == 0) {
    __hip_atomic_fetch_add(ctr, 1u, __ATOMIC_ACQ_REL, __HIP_MEMORY_SCOPE_AGENT);
    while (__hip_atomic_load(ctr, __ATOMIC_ACQUIRE, __HIP_MEMORY_SCOPE_AGENT) < target) {
      __builtin_amdgcn_s_sleep(2);
    }
  }
  __syncthreads();
  __threadfence();
}

// ---------------------------------------------------------------------------
// Kernel 2: persistent recurrence. 16 WGs x 4 waves. WG g owns output columns
// [g*64, g*64+64); its W_hh slice is TDM-DMA'd into LDS once (128 KB bf16).
// h_t goes to d_out in fp32 AND to a bf16 ping-pong buffer so the next step's
// A fragments need no conversion on the serial critical path.
// ---------------------------------------------------------------------------
__global__ __launch_bounds__(SEQ_BLOCK) void rnn_scan(float* __restrict__ out,
                                                      const unsigned short* __restrict__ WhhB,
                                                      unsigned short* __restrict__ hbuf,
                                                      unsigned* __restrict__ ctr) {
  extern __shared__ unsigned short lds_w[];     // 64 cols x 1024 K, bf16
  const int lane = threadIdx.x & 31;
  const int wave = threadIdx.x >> 5;
  const int half = lane >> 4;
  const int l16  = lane & 15;
  const int wgid = blockIdx.x;                  // 0..15 -> output col group
  const int nbase = wgid * 64;

  // Stage W_hh slice via the Tensor Data Mover: 64 rows x 1024 bf16 (128 KB)
  if (threadIdx.x < 32) {                       // one wave drives the TDM
    tdm_load_2d_bf16_to_lds(WhhB + (size_t)nbase * RNN_H, 0u, 64u, RNN_H, RNN_H);
    __builtin_amdgcn_s_wait_tensorcnt(0);
  }
  __syncthreads();

  // t = 0: h_0 = tanh(xh_0) in place; also seed bf16 parity-0 buffer.
  unsigned short* hb0 = hbuf;                                // parity 0
  for (int e = blockIdx.x * SEQ_BLOCK + threadIdx.x; e < RNN_N * RNN_H;
       e += GBLK * SEQ_BLOCK) {
    const int n = e >> 10, c = e & (RNN_H - 1);
    const size_t off = (size_t)n * (RNN_L * RNN_H) + c;
    const float v = tanhf(out[off]);
    out[off] = v;
    hb0[e] = f2bf(v);
  }
  unsigned phase = 1;
  grid_sync(ctr, GBLK * phase); ++phase;

  const int mt = wave;                          // batch tile: rows mt*16..+15
  const int nrow = mt * 16 + l16;               // batch row this lane gathers A from

  for (int t = 1; t < RNN_L; ++t) {
    v8f acc0 = {}, acc1 = {}, acc2 = {}, acc3 = {};
    const unsigned short* hprev = hbuf + (size_t)((t - 1) & 1) * (RNN_N * RNN_H)
                                       + (size_t)nrow * RNN_H;
    unsigned short* hcur = hbuf + (size_t)(t & 1) * (RNN_N * RNN_H);

    for (int k0 = 0; k0 < RNN_H; k0 += 32) {
      Frag a;
      fill_a_frag_bf16(a, hprev + k0 + 8 * half);

      Frag b0, b1, b2, b3;
      const unsigned short* pb = lds_w + (size_t)l16 * RNN_H + k0 + 8 * half;
      fill_b_frag_bf16(b0, pb);
      fill_b_frag_bf16(b1, pb + 16 * RNN_H);
      fill_b_frag_bf16(b2, pb + 32 * RNN_H);
      fill_b_frag_bf16(b3, pb + 48 * RNN_H);

      acc0 = __builtin_amdgcn_wmma_f32_16x16x32_bf16(false, a.v, false, b0.v, (short)0, acc0, false, false);
      acc1 = __builtin_amdgcn_wmma_f32_16x16x32_bf16(false, a.v, false, b1.v, (short)0, acc1, false, false);
      acc2 = __builtin_amdgcn_wmma_f32_16x16x32_bf16(false, a.v, false, b2.v, (short)0, acc2, false, false);
      acc3 = __builtin_amdgcn_wmma_f32_16x16x32_bf16(false, a.v, false, b3.v, (short)0, acc3, false, false);
    }

#pragma unroll
    for (int r = 0; r < 8; ++r) {
      const int mm = mt * 16 + r + 8 * half;    // batch index
      float* po = out + (size_t)mm * (RNN_L * RNN_H) + (size_t)t * RNN_H + nbase + l16;
      unsigned short* ph = hcur + (size_t)mm * RNN_H + nbase + l16;
      const float v0 = tanhf(acc0[r] + po[0]);
      const float v1 = tanhf(acc1[r] + po[16]);
      const float v2 = tanhf(acc2[r] + po[32]);
      const float v3 = tanhf(acc3[r] + po[48]);
      po[0]  = v0;  ph[0]  = f2bf(v0);
      po[16] = v1;  ph[16] = f2bf(v1);
      po[32] = v2;  ph[32] = f2bf(v2);
      po[48] = v3;  ph[48] = f2bf(v3);
    }
    grid_sync(ctr, GBLK * phase); ++phase;
  }
}

// ---------------------------------------------------------------------------
extern "C" void kernel_launch(void* const* d_in, const int* in_sizes, int n_in,
                              void* d_out, int out_size, void* d_ws, size_t ws_size,
                              hipStream_t stream) {
  const int*   x   = (const int*)  d_in[0];
  const float* emb = (const float*)d_in[1];
  const float* Whh = (const float*)d_in[2];
  const float* bhh = (const float*)d_in[3];
  const float* Wxh = (const float*)d_in[4];
  const float* bxh = (const float*)d_in[5];
  float* out = (float*)d_out;

  char* ws = (char*)d_ws;
  unsigned short* WxhB = (unsigned short*)(ws);                              // 2 MiB
  unsigned short* WhhB = (unsigned short*)(ws + (size_t)RNN_H * RNN_H * 2);  // 2 MiB
  float*          bias = (float*)(ws + (size_t)RNN_H * RNN_H * 4);           // 4 KiB
  unsigned*       ctr  = (unsigned*)(ws + (size_t)RNN_H * RNN_H * 4 + 4096);
  unsigned short* hbuf = (unsigned short*)(ws + (size_t)RNN_H * RNN_H * 4 + 8192); // 256 KiB

  prep_kernel<<<1024, 256, 0, stream>>>(Wxh, Whh, bxh, bhh, WxhB, WhhB, bias, ctr);

  dim3 g1(RNN_H / 64, (RNN_N * RNN_L) / 128);
  xh_gemm<<<g1, 256, 0, stream>>>(x, emb, WxhB, bias, out);

  const int ldsBytes = 64 * RNN_H * 2;   // 128 KiB per workgroup
  hipFuncSetAttribute((const void*)rnn_scan,
                      hipFuncAttributeMaxDynamicSharedMemorySize, ldsBytes);
  rnn_scan<<<GBLK, SEQ_BLOCK, ldsBytes, stream>>>(out, WhhB, hbuf, ctr);
}
